// LanguageModel_86758339379704
// MI455X (gfx1250) — matrix operand order repaired and checked by reading, not compiled, not run on hardware
//
#include <hip/hip_runtime.h>
#include <hip/hip_bf16.h>

// ---------------- model dims (reference constants) ----------------
#define LM_B 2
#define LM_S 2048
#define LM_D 768
#define LM_V 50257
#define LM_L 4
#define ROWS (LM_B * LM_S)   // 4096
#define TM 128
#define TN 128
#define TK 32
#define LSTRIDE 40           // bf16 elements per LDS row; 40*2 = 80 bytes (16B multiple)
#define FC_NPAD (((LM_V + TN - 1) / TN) * TN)   // 50304: padded row stride for bf16 fcW

typedef __attribute__((ext_vector_type(16))) __bf16 v16bf;
typedef __attribute__((ext_vector_type(8)))  float  v8f;

union FragBf { uint4 q[2]; v16bf v; };

__device__ __forceinline__ unsigned short f2bf(float f) {
  unsigned int u = __float_as_uint(f);
  u += 0x7FFFu + ((u >> 16) & 1u);       // round-to-nearest-even
  return (unsigned short)(u >> 16);
}

// ---- CDNA5 async global->LDS copy (ASYNCcnt-tracked), per-lane 16B ----
__device__ __forceinline__ void async_copy_b128(unsigned lds_byte_off,
                                                const void* gptr) {
  asm volatile("global_load_async_to_lds_b128 %0, %1, off"
               :: "v"(lds_byte_off), "v"((unsigned long long)(size_t)gptr)
               : "memory");
}
__device__ __forceinline__ void wait_async0() {
#if __has_builtin(__builtin_amdgcn_s_wait_asynccnt)
  __builtin_amdgcn_s_wait_asynccnt(0);
#else
  asm volatile("s_wait_asynccnt 0x0" ::: "memory");
#endif
}
// immediate-only builtin: keep args as non-type template params (constexpr)
template <int MASK, int N>
__device__ __forceinline__ void sched_group() {
#if __has_builtin(__builtin_amdgcn_sched_group_barrier)
  __builtin_amdgcn_sched_group_barrier(MASK, N, 0);
#endif
}

// ---------------- fp32 -> bf16 conversion ----------------
__global__ void cvt_bf16_kernel(const float* __restrict__ in,
                                unsigned short* __restrict__ out, size_t n) {
  size_t i = (size_t)blockIdx.x * blockDim.x + threadIdx.x;
  if (i < n) out[i] = f2bf(in[i]);
}

// fp32 [rows][N] -> bf16 [rows][Npad], zero-filled padding (keeps B rows 16B aligned)
__global__ void cvt_pad_bf16_kernel(const float* __restrict__ in,
                                    unsigned short* __restrict__ out,
                                    int rows, int N, int Npad) {
  size_t i = (size_t)blockIdx.x * blockDim.x + threadIdx.x;
  if (i >= (size_t)rows * Npad) return;
  int r = (int)(i / Npad), c = (int)(i % Npad);
  out[i] = (c < N) ? f2bf(in[(size_t)r * N + c]) : (unsigned short)0;
}

// ---------------- embedding gather -> bf16 ----------------
__global__ __launch_bounds__(256) void embed_kernel(const int* __restrict__ tokens,
                                                    const float* __restrict__ emb,
                                                    unsigned short* __restrict__ xb) {
  int row = blockIdx.x;                  // 0..ROWS-1
  int tok = tokens[row];
  const float* e = emb + (size_t)tok * LM_D;
  for (int i = threadIdx.x; i < LM_D; i += blockDim.x)
    xb[(size_t)row * LM_D + i] = f2bf(e[i]);
}

// ---------------- bf16 WMMA GEMM:  C[M,N] = A[M,K] * B[K,N] + bias ----------------
// A row-major bf16 (stride K), B row-major bf16 with row stride Bstride
// (Bstride % 8 == 0, columns >= N are zero), C fp32 (stride N).
// M % 128 == 0, K % 32 == 0. Double-buffered LDS; A staged via CDNA5 async
// global->LDS; B staged through VGPRs (transpose scatter) with global loads
// overlapped with the WMMA chain. Each wave owns a 32x64 output block
// (2 A-frags x 4 B-frags -> 8 WMMA, 12 LDS reads per chunk).
__global__ __launch_bounds__(256) void gemm_bf16_kernel(
    const unsigned short* __restrict__ A,
    const unsigned short* __restrict__ B,
    const float* __restrict__ bias,
    float* __restrict__ C,
    int M, int N, int K, int Bstride) {
  __shared__ __align__(16) unsigned short sA[2][TM * LSTRIDE];   // [m][k]
  __shared__ __align__(16) unsigned short sB[2][TN * LSTRIDE];   // transposed [n][k]

  const int tid  = threadIdx.x;
  const int wave = tid >> 5;             // 0..7 (wave32)
  const int lane = tid & 31;
  const int m0 = blockIdx.y * TM;
  const int n0 = blockIdx.x * TN;

  const int aL = lane & 15;              // M (or N) index within 16x16 tile
  const int g  = lane >> 4;              // K-half group
  const int wr = wave & 3;               // wave's 32-row band (0..3)
  const int wc = wave >> 2;              // wave's 64-col band (0..1)

  // staging coordinates
  const int arow = tid >> 2;             // 0..63 (+64 on second pass)
  const int acol = (tid & 3) * 8;        // 0,8,16,24
  const int bk   = tid >> 4;             // 0..15 (+16 on second pass)
  const int bn   = (tid & 15) * 8;       // 0..120

  v8f acc[2][4] = {};
  uint4 btmp[2];
  const int nChunks = K / TK;

  // ---- issue async A-tile staging for chunk c into buffer buf ----
  auto issueA = [&](int c, int buf) {
    const unsigned short* Ab = A + (size_t)m0 * K + c * TK;
    #pragma unroll
    for (int it = 0; it < 2; ++it) {
      int row = it * 64 + arow;
      unsigned loff = (unsigned)(size_t)&sA[buf][row * LSTRIDE + acol];
      async_copy_b128(loff, Ab + (size_t)row * K + acol);
    }
  };
  // ---- pull B rows for chunk c into registers (always aligned b128) ----
  auto loadB = [&](int c) {
    const unsigned short* Bb = B + (size_t)(c * TK) * Bstride + n0 + bn;
    #pragma unroll
    for (int it = 0; it < 2; ++it)
      btmp[it] = *(const uint4*)(Bb + (size_t)(it * 16 + bk) * Bstride);
  };
  // ---- scatter B registers transposed into LDS buffer buf ----
  auto commitB = [&](int buf) {
    #pragma unroll
    for (int it = 0; it < 2; ++it) {
      int kk = it * 16 + bk;
      const unsigned short* t = (const unsigned short*)&btmp[it];
      #pragma unroll
      for (int i = 0; i < 8; ++i)
        sB[buf][(bn + i) * LSTRIDE + kk] = t[i];
    }
  };
  // ---- 8 WMMA tiles for this wave's 32x64 block ----
  auto compute = [&](int buf) {
    FragBf fa[2], fb[4];
    const unsigned short* ap = &sA[buf][(wr * 32 + aL) * LSTRIDE];
    #pragma unroll
    for (int i = 0; i < 2; ++i) {
      const unsigned short* api = ap + i * 16 * LSTRIDE;
      fa[i].q[0] = *(const uint4*)(api + 8 * g);        // K = 8g..8g+7
      fa[i].q[1] = *(const uint4*)(api + 16 + 8 * g);   // K = 16+8g..
    }
    const unsigned short* bbase = &sB[buf][(size_t)(wc * 64 + aL) * LSTRIDE];
    #pragma unroll
    for (int j = 0; j < 4; ++j) {
      const unsigned short* bp = bbase + (size_t)j * 16 * LSTRIDE;
      fb[j].q[0] = *(const uint4*)(bp + 8 * g);         // K = 8g..8g+7
      fb[j].q[1] = *(const uint4*)(bp + 16 + 8 * g);    // K = 16+8g..
    }
    #pragma unroll
    for (int j = 0; j < 4; ++j) {
      acc[0][j] = __builtin_amdgcn_wmma_f32_16x16x32_bf16(
          false, fa[0].v, false, fb[j].v, (short)0, acc[0][j], false, false);
      acc[1][j] = __builtin_amdgcn_wmma_f32_16x16x32_bf16(
          false, fa[1].v, false, fb[j].v, (short)0, acc[1][j], false, false);
    }
    // pipeline ladder: 6 DS reads (fa0,fa1,fb0), then (2 wmma, 2 DS reads)x3,
    // then final 2 wmma. 0x100 = DS-read, 0x8 = MFMA/WMMA.
    sched_group<0x100, 6>();
    sched_group<0x008, 2>();
    sched_group<0x100, 2>();
    sched_group<0x008, 2>();
    sched_group<0x100, 2>();
    sched_group<0x008, 2>();
    sched_group<0x100, 2>();
    sched_group<0x008, 2>();
  };

  // prologue: stage chunk 0 into buffer 0
  issueA(0, 0);
  loadB(0);
  commitB(0);
  wait_async0();
  __syncthreads();

  for (int c = 0; c < nChunks; ++c) {
    const int buf = c & 1;
    const int nbuf = buf ^ 1;
    if (c + 1 < nChunks) {
      issueA(c + 1, nbuf);     // async engine fills next A tile
      loadB(c + 1);            // global B loads in flight during compute
    }
    compute(buf);
    if (c + 1 < nChunks) commitB(nbuf);
    wait_async0();
    __syncthreads();
  }

  // ---- store: VGPR r holds M = 8*(lane>>4) + r, N = lane&15 ----
  #pragma unroll
  for (int i = 0; i < 2; ++i) {
    #pragma unroll
    for (int j = 0; j < 4; ++j) {
      int col = n0 + wc * 64 + j * 16 + aL;
      if (col < N) {
        float bv = bias[col];
        #pragma unroll
        for (int r = 0; r < 8; ++r) {
          int row = m0 + wr * 32 + i * 16 + g * 8 + r;
          // non-temporal: keep the 192MB L2 for the (bf16) B matrix
          __builtin_nontemporal_store(acc[i][j][r] + bv, &C[(size_t)row * N + col]);
        }
      }
    }
  }
}

// ---------------- minGRU sequential scan (exact recurrence of the log-space scan) ----
// h_t = (1-z_t) h_{t-1} + z_t * g(hp_t); z = sigmoid(k); g(x) = x+0.5 (x>=0) else sigmoid(x)
__global__ __launch_bounds__(256) void scan_kernel(const float* __restrict__ kbuf,
                                                   const float* __restrict__ hbuf,
                                                   float* __restrict__ xnew) {
  int c = blockIdx.x * blockDim.x + threadIdx.x;     // channel 0..B*D-1
  if (c >= LM_B * LM_D) return;
  int b = c / LM_D, d = c % LM_D;
  const float* kp = kbuf + (size_t)b * LM_S * LM_D + d;
  const float* hp = hbuf + (size_t)b * LM_S * LM_D + d;
  float* op       = xnew + (size_t)b * LM_S * LM_D + d;
  float h = 0.5f;                                    // g(h0 = 0)
  for (int t = 0; t < LM_S; ++t) {
    float kv = kp[(size_t)t * LM_D];
    float hv = hp[(size_t)t * LM_D];
    float z  = 1.0f / (1.0f + __expf(-kv));
    float gh = (hv >= 0.0f) ? (hv + 0.5f) : (1.0f / (1.0f + __expf(-hv)));
    h = (1.0f - z) * h + z * gh;
    op[(size_t)t * LM_D] = h;
  }
}

// ---------------- LayerNorm over D, writes bf16 activations ----------------
__global__ __launch_bounds__(256) void ln_kernel(const float* __restrict__ xin,
                                                 const float* __restrict__ gw,
                                                 const float* __restrict__ bw,
                                                 unsigned short* __restrict__ xb) {
  int row = blockIdx.x;
  const float* xr = xin + (size_t)row * LM_D;
  float s = 0.0f, s2 = 0.0f;
  for (int i = threadIdx.x; i < LM_D; i += blockDim.x) {
    float v = xr[i]; s += v; s2 += v * v;
  }
  #pragma unroll
  for (int off = 16; off > 0; off >>= 1) {
    s  += __shfl_down(s,  off, 32);
    s2 += __shfl_down(s2, off, 32);
  }
  __shared__ float shs[8], shs2[8];
  __shared__ float mu_s, rstd_s;
  int wid = threadIdx.x >> 5, lid = threadIdx.x & 31;
  if (lid == 0) { shs[wid] = s; shs2[wid] = s2; }
  __syncthreads();
  if (threadIdx.x == 0) {
    float S1 = 0.0f, S2 = 0.0f;
    #pragma unroll
    for (int w = 0; w < 8; ++w) { S1 += shs[w]; S2 += shs2[w]; }
    float mu  = S1 / (float)LM_D;
    float var = S2 / (float)LM_D - mu * mu;
    mu_s = mu; rstd_s = rsqrtf(var + 1e-5f);
  }
  __syncthreads();
  float mu = mu_s, rstd = rstd_s;
  for (int i = threadIdx.x; i < LM_D; i += blockDim.x) {
    float v = (xr[i] - mu) * rstd * gw[i] + bw[i];
    xb[(size_t)row * LM_D + i] = f2bf(v);
  }
}

// ---------------- launch ----------------
extern "C" void kernel_launch(void* const* d_in, const int* in_sizes, int n_in,
                              void* d_out, int out_size, void* d_ws, size_t ws_size,
                              hipStream_t stream) {
  const int*   tokens = (const int*)d_in[0];
  const float* emb    = (const float*)d_in[1];
  const float* Wz     = (const float*)d_in[2];
  const float* bz     = (const float*)d_in[3];
  const float* Wh     = (const float*)d_in[4];
  const float* bh     = (const float*)d_in[5];
  const float* ln_g   = (const float*)d_in[6];
  const float* ln_b   = (const float*)d_in[7];
  const float* fcW    = (const float*)d_in[8];
  const float* fcb    = (const float*)d_in[9];
  float* out = (float*)d_out;

  char* ws = (char*)d_ws;
  size_t off = 0;
  auto wsalloc = [&](size_t bytes) -> char* {
    char* p = ws + off;
    off += (bytes + 255) & ~(size_t)255;
    return p;
  };
  unsigned short* xb   = (unsigned short*)wsalloc((size_t)ROWS * LM_D * 2);
  float*          kbuf = (float*)wsalloc((size_t)ROWS * LM_D * 4);
  float*          hbuf = (float*)wsalloc((size_t)ROWS * LM_D * 4);
  float*          xnew = (float*)wsalloc((size_t)ROWS * LM_D * 4);
  unsigned short* WzB  = (unsigned short*)wsalloc((size_t)LM_L * LM_D * LM_D * 2);
  unsigned short* WhB  = (unsigned short*)wsalloc((size_t)LM_L * LM_D * LM_D * 2);
  unsigned short* fcWB = (unsigned short*)wsalloc((size_t)LM_D * FC_NPAD * 2);
  if (off > ws_size) return;   // workspace too small; nothing safe to do

  // weights -> bf16 (every call; deterministic). fcW gets a padded row stride.
  {
    size_t n  = (size_t)LM_L * LM_D * LM_D;
    size_t nf = (size_t)LM_D * FC_NPAD;
    cvt_bf16_kernel<<<(unsigned)((n + 255) / 256), 256, 0, stream>>>(Wz, WzB, n);
    cvt_bf16_kernel<<<(unsigned)((n + 255) / 256), 256, 0, stream>>>(Wh, WhB, n);
    cvt_pad_bf16_kernel<<<(unsigned)((nf + 255) / 256), 256, 0, stream>>>(
        fcW, fcWB, LM_D, LM_V, FC_NPAD);
  }

  embed_kernel<<<ROWS, 256, 0, stream>>>(tokens, emb, xb);

  dim3 gsmall((LM_D + TN - 1) / TN, ROWS / TM);
  for (int l = 0; l < LM_L; ++l) {
    gemm_bf16_kernel<<<gsmall, 256, 0, stream>>>(
        xb, WzB + (size_t)l * LM_D * LM_D, bz + (size_t)l * LM_D, kbuf,
        ROWS, LM_D, LM_D, LM_D);
    gemm_bf16_kernel<<<gsmall, 256, 0, stream>>>(
        xb, WhB + (size_t)l * LM_D * LM_D, bh + (size_t)l * LM_D, hbuf,
        ROWS, LM_D, LM_D, LM_D);
    scan_kernel<<<(LM_B * LM_D + 255) / 256, 256, 0, stream>>>(kbuf, hbuf, xnew);
    ln_kernel<<<ROWS, 256, 0, stream>>>(xnew, ln_g, ln_b, xb);
  }

  dim3 gbig((LM_V + TN - 1) / TN, ROWS / TM);
  gemm_bf16_kernel<<<gbig, 256, 0, stream>>>(xb, fcWB, fcb, out,
                                             ROWS, LM_V, LM_D, FC_NPAD);
}
